// SwinTransformerBlock_60825326846684
// MI455X (gfx1250) — compile-verified, matrix operand order checked by reading, not run on hardware
//
#include <hip/hip_runtime.h>
#include <hip/hip_bf16.h>
#include <math.h>

// ---------------------------------------------------------------------------
// Swin block for MI455X (gfx1250, wave32). All GEMM work on
// v_wmma_f32_16x16x32_f16 (16x16 tiles, K-step 32, f32 accumulate).
// GEMM uses 2Mx2N register blocking per wave: 4 WMMAs per 8 b128 loads.
// ---------------------------------------------------------------------------

typedef _Float16 half8  __attribute__((ext_vector_type(8)));
typedef _Float16 half16 __attribute__((ext_vector_type(16)));
typedef float    float8 __attribute__((ext_vector_type(8)));

#define BATCH   32
#define HW      56
#define C_DIM   384
#define WS_     7
#define N_WIN   49          // tokens per window
#define HEADS   12
#define HDIM    32
#define ROWS    (BATCH*HW*HW)     // 100352 tokens
#define NWIN_T  (BATCH*64)        // 2048 windows

// ---- WMMA fragment load (16-bit A/B layout, ISA 7.12.2) -------------------
// lane = lo + 16*hi.  halves 0..7  : K = kbase + 8*hi + i
//                     halves 8..15 : K = kbase + 16 + 8*hi + i
__device__ __forceinline__ half16 load_frag(const _Float16* __restrict__ p, int hi) {
    const half8 a0 = *(const half8*)(p + 8 * hi);
    const half8 a1 = *(const half8*)(p + 16 + 8 * hi);
    half16 r;
#pragma unroll
    for (int i = 0; i < 8; ++i) { r[i] = a0[i]; r[8 + i] = a1[i]; }
    return r;
}

__device__ __forceinline__ half16 zero_frag() {
    half16 r = {};
    return r;
}

// ---------------------------------------------------------------------------
// fp32 -> f16 weight conversion
// ---------------------------------------------------------------------------
__global__ void cvt_f16_kernel(const float* __restrict__ src, _Float16* __restrict__ dst, int n) {
    int i = blockIdx.x * blockDim.x + threadIdx.x;
    if (i < n) dst[i] = (_Float16)src[i];
}

// ---------------------------------------------------------------------------
// LayerNorm (one wave per 384-wide row).
// mode 0: plain rows   src[row*384 + c]            -> dst[row*384 + c]
// mode 1: LN1 fused with roll(-3,-3)+window partition: output row o = w*49+n,
//         source pixel = image coord of shifted/partitioned token.
// ---------------------------------------------------------------------------
__global__ __launch_bounds__(256)
void ln_kernel(const float* __restrict__ src, const float* __restrict__ g,
               const float* __restrict__ bb, _Float16* __restrict__ dst, int mode) {
    const int lane = threadIdx.x & 31;
    const int wave = threadIdx.x >> 5;
    const int o = blockIdx.x * 8 + wave;     // output row, < ROWS
    size_t srow;
    if (mode == 1) {
        int w = o / N_WIN, n = o % N_WIN;
        int b = w >> 6, wi = w & 63;
        int wh = wi >> 3, ww = wi & 7;
        int r = n / WS_, c = n % WS_;
        int y = wh * WS_ + r, x = ww * WS_ + c;
        int yi = y + 3; if (yi >= HW) yi -= HW;   // roll(-3): h[y] = x[(y+3)%56]
        int xi = x + 3; if (xi >= HW) xi -= HW;
        srow = ((size_t)((b * HW + yi) * HW + xi)) * C_DIM;
    } else {
        srow = (size_t)o * C_DIM;
    }
    float v[12];
    float s = 0.f;
#pragma unroll
    for (int j = 0; j < 12; ++j) { v[j] = src[srow + lane + 32 * j]; s += v[j]; }
#pragma unroll
    for (int off = 16; off; off >>= 1) s += __shfl_xor(s, off, 32);
    const float mu = s * (1.f / 384.f);
    float q = 0.f;
#pragma unroll
    for (int j = 0; j < 12; ++j) { float d = v[j] - mu; q += d * d; }
#pragma unroll
    for (int off = 16; off; off >>= 1) q += __shfl_xor(q, off, 32);
    const float rstd = rsqrtf(q * (1.f / 384.f) + 1e-5f);
#pragma unroll
    for (int j = 0; j < 12; ++j) {
        int c = lane + 32 * j;
        dst[(size_t)o * C_DIM + c] = (_Float16)((v[j] - mu) * rstd * g[c] + bb[c]);
    }
}

// ---------------------------------------------------------------------------
// WMMA GEMM:  out[M,N] = A[M,K] (f16, row-major) * W[N,K]^T + bias
// 256 threads = 8 waves; each wave owns a 32x32 register tile (2Mx2N WMMA
// tiles -> each fragment feeds 2 WMMAs). Block = 2x4 waves = 64x128 output.
// mode 0: store f16                    -> out16
// mode 1: GELU(exact) then store f16   -> out16
// mode 2: proj: window-reverse + roll(+3,+3) + residual(addend=x) -> out32 (image layout)
// mode 3: residual: out32 = addend + acc (row-major)
// ---------------------------------------------------------------------------
__global__ __launch_bounds__(256)
void gemm_wmma_kernel(const _Float16* __restrict__ A, const _Float16* __restrict__ W,
                      const float* __restrict__ bias, int K, int N, int mode,
                      _Float16* __restrict__ out16, float* __restrict__ out32,
                      const float* __restrict__ addend) {
    const int lane = threadIdx.x & 31;
    const int wave = threadIdx.x >> 5;
    const int lo = lane & 15, hi = lane >> 4;
    const int row0 = blockIdx.y * 64 + (wave >> 2) * 32;   // 2 waves in M
    const int col0 = blockIdx.x * 128 + (wave & 3) * 32;   // 4 waves in N

    const _Float16* __restrict__ ar0 = A + (size_t)(row0 + lo) * K;
    const _Float16* __restrict__ ar1 = A + (size_t)(row0 + 16 + lo) * K;
    const _Float16* __restrict__ wr0 = W + (size_t)(col0 + lo) * K;
    const _Float16* __restrict__ wr1 = W + (size_t)(col0 + 16 + lo) * K;

    float8 acc00 = {}, acc01 = {}, acc10 = {}, acc11 = {};
    for (int k = 0; k < K; k += 32) {
        half16 a0 = load_frag(ar0 + k, hi);
        half16 a1 = load_frag(ar1 + k, hi);
        half16 b0 = load_frag(wr0 + k, hi);
        half16 b1 = load_frag(wr1 + k, hi);
        __builtin_prefetch(ar0 + k + 64, 0, 1);   // global_prefetch_b8
        __builtin_prefetch(ar1 + k + 64, 0, 1);
        __builtin_prefetch(wr0 + k + 64, 0, 1);
        __builtin_prefetch(wr1 + k + 64, 0, 1);
        acc00 = __builtin_amdgcn_wmma_f32_16x16x32_f16(false, a0, false, b0, (short)0, acc00, false, false);
        acc01 = __builtin_amdgcn_wmma_f32_16x16x32_f16(false, a0, false, b1, (short)0, acc01, false, false);
        acc10 = __builtin_amdgcn_wmma_f32_16x16x32_f16(false, a1, false, b0, (short)0, acc10, false, false);
        acc11 = __builtin_amdgcn_wmma_f32_16x16x32_f16(false, a1, false, b1, (short)0, acc11, false, false);
    }

#pragma unroll
    for (int sm = 0; sm < 2; ++sm) {
#pragma unroll
        for (int sn = 0; sn < 2; ++sn) {
            const float8& acc = (sm == 0) ? ((sn == 0) ? acc00 : acc01)
                                          : ((sn == 0) ? acc10 : acc11);
            const int col = col0 + sn * 16 + lo;
            const float bs = bias[col];
#pragma unroll
            for (int v = 0; v < 8; ++v) {
                const int m = row0 + sm * 16 + v + 8 * hi;  // C layout: VGPR v -> M = v+8*hi
                float val = acc[v] + bs;
                if (mode == 0) {
                    out16[(size_t)m * N + col] = (_Float16)val;
                } else if (mode == 1) {
                    val = 0.5f * val * (1.f + erff(val * 0.70710678118654752f));
                    out16[(size_t)m * N + col] = (_Float16)val;
                } else if (mode == 2) {
                    int w = m / N_WIN, n = m % N_WIN;
                    int b = w >> 6, wi = w & 63;
                    int wh = wi >> 3, ww = wi & 7;
                    int r = n / WS_, c = n % WS_;
                    int y = wh * WS_ + r, x = ww * WS_ + c;
                    int yo = y + 3; if (yo >= HW) yo -= HW;   // roll(+3)
                    int xo = x + 3; if (xo >= HW) xo -= HW;
                    size_t idx = ((size_t)((b * HW + yo) * HW + xo)) * C_DIM + col;
                    out32[idx] = addend[idx] + val;
                } else { // mode 3
                    size_t idx = (size_t)m * N + col;
                    out32[idx] = addend[idx] + val;
                }
            }
        }
    }
}

// ---------------------------------------------------------------------------
// Fused windowed attention, one wave per (window, head).
//   S = (Q*scale) K^T + rel_bias + shift_mask ; P = softmax(S) ; O = P V
// Q,K,V read from qkv buffer (row stride 1152 = [q|k|v] x 12 heads x 32).
// 49 tokens padded to 64 -> 4x4 S tiles + 4x2 O tiles = 32 WMMAs / wave.
// ---------------------------------------------------------------------------
__device__ __forceinline__ int region1d(int y) { return (y < 49) ? 0 : ((y < 53) ? 1 : 2); }

__global__ __launch_bounds__(32)
void attn_kernel(const _Float16* __restrict__ qkv,
                 const float* __restrict__ bias_table,   // (169, 12)
                 _Float16* __restrict__ out) {           // (rows, 384) f16
    __shared__ float    S[64 * 64];
    __shared__ _Float16 P[64 * 64];
    __shared__ _Float16 Vt[32 * 64];

    const int w = blockIdx.x;
    const int h = blockIdx.y;
    const int lane = threadIdx.x;
    const int lo = lane & 15, hi = lane >> 4;
    const int wi = w & 63;
    const int wh = wi >> 3, ww = wi & 7;
    const size_t qbase = (size_t)w * N_WIN * 1152 + h * HDIM;

    // ---- Q, K fragments (A: 16x32 rows=query ; B: 32x16 cols=key) ----
    half16 qf[4], kf[4];
#pragma unroll
    for (int t = 0; t < 4; ++t) {
        int m = t * 16 + lo;
        qf[t] = (m < N_WIN) ? load_frag(qkv + qbase + (size_t)m * 1152, hi) : zero_frag();
        kf[t] = (m < N_WIN) ? load_frag(qkv + qbase + 384 + (size_t)m * 1152, hi) : zero_frag();
    }
    // ---- S = Q K^T ----
#pragma unroll
    for (int mt = 0; mt < 4; ++mt) {
#pragma unroll
        for (int nt = 0; nt < 4; ++nt) {
            float8 acc = {};
            acc = __builtin_amdgcn_wmma_f32_16x16x32_f16(false, qf[mt], false, kf[nt],
                                                         (short)0, acc, false, false);
#pragma unroll
            for (int v = 0; v < 8; ++v)
                S[(mt * 16 + v + 8 * hi) * 64 + nt * 16 + lo] = acc[v];
        }
    }

    // ---- stage V transposed: Vt[d][key] ----
    for (int key = lane; key < 64; key += 32) {
        if (key < N_WIN) {
            const _Float16* vp = qkv + qbase + 768 + (size_t)key * 1152;
#pragma unroll
            for (int d = 0; d < 32; ++d) Vt[d * 64 + key] = vp[d];
        } else {
#pragma unroll
            for (int d = 0; d < 32; ++d) Vt[d * 64 + key] = (_Float16)0.f;
        }
    }
    __syncthreads();

    // ---- softmax (+ relative bias, + shifted-window mask), 2 rows per lane ----
    const float scale = 0.17677669529663687f;   // 1/sqrt(32)
    for (int m = lane; m < 64; m += 32) {
        if (m < N_WIN) {
            const int r1 = m / WS_, c1 = m % WS_;
            const int reg1 = region1d(wh * WS_ + r1) * 3 + region1d(ww * WS_ + c1);
            float mx = -1e30f;
            for (int j = 0; j < N_WIN; ++j) {
                const int r2 = j / WS_, c2 = j % WS_;
                const int idx = (r1 - r2 + 6) * 13 + (c1 - c2 + 6);
                const int reg2 = region1d(wh * WS_ + r2) * 3 + region1d(ww * WS_ + c2);
                float val = S[m * 64 + j] * scale + bias_table[idx * HEADS + h]
                          + ((reg1 == reg2) ? 0.f : -100.f);
                S[m * 64 + j] = val;
                mx = fmaxf(mx, val);
            }
            float sum = 0.f;
            for (int j = 0; j < N_WIN; ++j) {
                float e = __expf(S[m * 64 + j] - mx);
                S[m * 64 + j] = e;
                sum += e;
            }
            const float inv = 1.f / sum;
            for (int j = 0; j < N_WIN; ++j) P[m * 64 + j] = (_Float16)(S[m * 64 + j] * inv);
            for (int j = N_WIN; j < 64; ++j) P[m * 64 + j] = (_Float16)0.f;
        } else {
            for (int j = 0; j < 64; ++j) P[m * 64 + j] = (_Float16)0.f;
        }
    }
    __syncthreads();

    // ---- O = P V :  (64x64)x(64x32), B fragments from Vt (col = dim) ----
    half16 vb[2][2];
#pragma unroll
    for (int nd = 0; nd < 2; ++nd)
#pragma unroll
        for (int ks = 0; ks < 2; ++ks)
            vb[nd][ks] = load_frag(&Vt[(nd * 16 + lo) * 64 + 32 * ks], hi);

#pragma unroll
    for (int mt = 0; mt < 4; ++mt) {
        half16 pf0 = load_frag(&P[(mt * 16 + lo) * 64 + 0], hi);
        half16 pf1 = load_frag(&P[(mt * 16 + lo) * 64 + 32], hi);
#pragma unroll
        for (int nd = 0; nd < 2; ++nd) {
            float8 acc = {};
            acc = __builtin_amdgcn_wmma_f32_16x16x32_f16(false, pf0, false, vb[nd][0],
                                                         (short)0, acc, false, false);
            acc = __builtin_amdgcn_wmma_f32_16x16x32_f16(false, pf1, false, vb[nd][1],
                                                         (short)0, acc, false, false);
#pragma unroll
            for (int v = 0; v < 8; ++v) {
                const int m = mt * 16 + v + 8 * hi;
                if (m < N_WIN)
                    out[((size_t)w * N_WIN + m) * C_DIM + h * HDIM + nd * 16 + lo] =
                        (_Float16)acc[v];
            }
        }
    }
}

// ---------------------------------------------------------------------------
// Host-side launcher
// ---------------------------------------------------------------------------
extern "C" void kernel_launch(void* const* d_in, const int* in_sizes, int n_in,
                              void* d_out, int out_size, void* d_ws, size_t ws_size,
                              hipStream_t stream) {
    (void)in_sizes; (void)n_in; (void)out_size; (void)ws_size;

    const float* x        = (const float*)d_in[0];
    const float* norm1_w  = (const float*)d_in[1];
    const float* norm1_b  = (const float*)d_in[2];
    const float* qkv_w    = (const float*)d_in[3];
    const float* qkv_b    = (const float*)d_in[4];
    const float* rel_tab  = (const float*)d_in[5];
    const float* proj_w   = (const float*)d_in[6];
    const float* proj_b   = (const float*)d_in[7];
    const float* norm2_w  = (const float*)d_in[8];
    const float* norm2_b  = (const float*)d_in[9];
    const float* mlp1_w   = (const float*)d_in[10];
    const float* mlp1_b   = (const float*)d_in[11];
    const float* mlp2_w   = (const float*)d_in[12];
    const float* mlp2_b   = (const float*)d_in[13];
    float* out = (float*)d_out;

    char* p = (char*)d_ws;
    auto alloc = [&](size_t bytes) -> void* {
        void* r = (void*)p;
        p += (bytes + 255) & ~(size_t)255;
        return r;
    };
    _Float16* qkvw16 = (_Float16*)alloc((size_t)1152 * 384 * 2);
    _Float16* projw16= (_Float16*)alloc((size_t)384 * 384 * 2);
    _Float16* m1w16  = (_Float16*)alloc((size_t)1536 * 384 * 2);
    _Float16* m2w16  = (_Float16*)alloc((size_t)384 * 1536 * 2);
    _Float16* xw16   = (_Float16*)alloc((size_t)ROWS * C_DIM * 2);
    _Float16* qkv16  = (_Float16*)alloc((size_t)ROWS * 1152 * 2);
    _Float16* attn16 = (_Float16*)alloc((size_t)ROWS * C_DIM * 2);
    float*    x1     = (float*)  alloc((size_t)ROWS * C_DIM * 4);
    _Float16* h16    = (_Float16*)alloc((size_t)ROWS * C_DIM * 2);
    _Float16* mid16  = (_Float16*)alloc((size_t)ROWS * 1536 * 2);

    // 1) weights -> f16
    auto cvt = [&](const float* s, _Float16* d, int n) {
        cvt_f16_kernel<<<(n + 255) / 256, 256, 0, stream>>>(s, d, n);
    };
    cvt(qkv_w,  qkvw16, 1152 * 384);
    cvt(proj_w, projw16, 384 * 384);
    cvt(mlp1_w, m1w16,  1536 * 384);
    cvt(mlp2_w, m2w16,  384 * 1536);

    // 2) LN1 + roll(-3,-3) + window partition -> xw16 (rows = window tokens)
    ln_kernel<<<ROWS / 8, 256, 0, stream>>>(x, norm1_w, norm1_b, xw16, 1);

    // 3) QKV GEMM: (100352 x 384) x (1152 x 384)^T  -> qkv16
    gemm_wmma_kernel<<<dim3(1152 / 128, ROWS / 64), 256, 0, stream>>>(
        xw16, qkvw16, qkv_b, 384, 1152, 0, qkv16, nullptr, nullptr);

    // 4) fused attention per (window, head) -> attn16
    attn_kernel<<<dim3(NWIN_T, HEADS), 32, 0, stream>>>(qkv16, rel_tab, attn16);

    // 5) proj GEMM fused with window-reverse + roll(+3,+3) + residual -> x1 (f32)
    gemm_wmma_kernel<<<dim3(384 / 128, ROWS / 64), 256, 0, stream>>>(
        attn16, projw16, proj_b, 384, 384, 2, nullptr, x1, x);

    // 6) LN2 -> h16
    ln_kernel<<<ROWS / 8, 256, 0, stream>>>(x1, norm2_w, norm2_b, h16, 0);

    // 7) MLP1 + exact GELU -> mid16
    gemm_wmma_kernel<<<dim3(1536 / 128, ROWS / 64), 256, 0, stream>>>(
        h16, m1w16, mlp1_b, 384, 1536, 1, mid16, nullptr, nullptr);

    // 8) MLP2 + residual(x1) -> d_out (f32)
    gemm_wmma_kernel<<<dim3(384 / 128, ROWS / 64), 256, 0, stream>>>(
        mid16, m2w16, mlp2_b, 1536, 384, 3, nullptr, out, x1);
}
